// MultiHeadSelfAttention_833223655758
// MI455X (gfx1250) — matrix operand order repaired and checked by reading, not compiled, not run on hardware
//
#include <hip/hip_runtime.h>
#include <hip/hip_bf16.h>

// ---------------------------------------------------------------------------
// GPT block forward for MI455X (gfx1250, wave32, WMMA).
// bf16 inputs to v_wmma_f32_16x16x32_bf16, f32 accumulate, fp32 residual.
// Weights pre-transposed to [N,K] at conversion so both GEMM tiles stage as
// row-major async copies (GLOBAL_LOAD_ASYNC_TO_LDS_B128, ASYNCcnt), double
// buffered so WMMA overlaps the incoming tile. Flash attention streams KV in
// 64-wide tiles with online softmax (never materializes [B,H,T,T]).
// ---------------------------------------------------------------------------

typedef unsigned short u16;
typedef __attribute__((ext_vector_type(16))) __bf16 v16bf;
typedef __attribute__((ext_vector_type(8)))  float  v8f;
typedef int v4i __attribute__((vector_size(16)));   // matches builtin param

#define AS1 __attribute__((address_space(1)))
#define AS3 __attribute__((address_space(3)))

#if __has_builtin(__builtin_amdgcn_global_load_async_to_lds_b128)
#define HAS_ASYNC_LDS 1
#else
#define HAS_ASYNC_LDS 0
#endif

#if HAS_ASYNC_LDS
#if __has_builtin(__builtin_amdgcn_s_wait_asynccnt)
#define WAIT_ASYNC(n) __builtin_amdgcn_s_wait_asynccnt(n)
#else
#define WAIT_ASYNC(n) asm volatile("s_wait_asynccnt %0" ::"n"(n) : "memory")
#endif
#else
#define WAIT_ASYNC(n) ((void)0)
#endif

constexpr int kB  = 8;
constexpr int kT  = 1024;
constexpr int kD  = 1024;
constexpr int kH  = 16;
constexpr int kHD = 64;
constexpr int kL  = 4;
constexpr int kDF = 4096;
constexpr int kBT = kB * kT;

// ---- helpers --------------------------------------------------------------

__device__ __forceinline__ u16 f2bf(float f) {
  unsigned u = __float_as_uint(f);
  return (u16)((u + 0x7FFFu + ((u >> 16) & 1u)) >> 16);
}

// 16-byte copy global -> LDS. Async (no VGPR round trip) when available.
__device__ __forceinline__ void cp16(u16* ldst, const u16* gsrc) {
#if HAS_ASYNC_LDS
  __builtin_amdgcn_global_load_async_to_lds_b128((AS1 v4i*)gsrc,
                                                 (AS3 v4i*)ldst, 0, 0);
#else
  *(uint4*)ldst = *(const uint4*)gsrc;
#endif
}

union FragBF {
  uint4  q[2];
  u16    s[16];
  v16bf  v;
};

__device__ __forceinline__ v16bf frag_ld(const u16* base, int off0, int off1) {
  FragBF f;
  f.q[0] = *(const uint4*)(base + off0);
  f.q[1] = *(const uint4*)(base + off1);
  return f.v;
}

// A fragment (16x32 bf16), ISA 7.12.2: lanes 0-15 hold K {0..7,16..23}.
__device__ __forceinline__ v16bf frag_a(const u16* lds, int row, int ldk,
                                        int lane, int koff) {
  int k0 = koff + ((lane & 16) ? 8 : 0);
  return frag_ld(lds, row * ldk + k0, row * ldk + k0 + 16);
}

// B fragment (32x16 bf16) from N-major tile: lanes 0-15 hold K {0..7,16..23}.
__device__ __forceinline__ v16bf frag_b(const u16* lds, int nrow, int ldk,
                                        int lane, int koff) {
  int k0 = koff + ((lane & 16) ? 8 : 0);
  return frag_ld(lds, nrow * ldk + k0, nrow * ldk + k0 + 16);
}

__device__ __forceinline__ v8f wmma_bf16(v16bf a, v16bf b, v8f c) {
  return __builtin_amdgcn_wmma_f32_16x16x32_bf16(false, a, false, b, (short)0,
                                                 c, false, false);
}

__device__ __forceinline__ v8f v8f_zero() {
  v8f z;
#pragma unroll
  for (int e = 0; e < 8; ++e) z[e] = 0.f;
  return z;
}

__device__ __forceinline__ float redmax16(float v) {
  v = fmaxf(v, __shfl_xor(v, 1, 32));
  v = fmaxf(v, __shfl_xor(v, 2, 32));
  v = fmaxf(v, __shfl_xor(v, 4, 32));
  v = fmaxf(v, __shfl_xor(v, 8, 32));
  return v;
}

__device__ __forceinline__ float redsum16(float v) {
  v += __shfl_xor(v, 1, 32);
  v += __shfl_xor(v, 2, 32);
  v += __shfl_xor(v, 4, 32);
  v += __shfl_xor(v, 8, 32);
  return v;
}

// ---- weight conversion: f32 [L,K,N] -> bf16 [L,N,K] (transposed) ----------

__global__ void convert_transpose_kernel(const float* __restrict__ src,
                                         u16* __restrict__ dst, int Kd, int Nd,
                                         long total) {
  long stride = (long)gridDim.x * blockDim.x;
  for (long i = (long)blockIdx.x * blockDim.x + threadIdx.x; i < total;
       i += stride) {
    long plane = (long)Kd * Nd;
    long l = i / plane;
    long rem = i - l * plane;
    int k = (int)(rem / Nd), n = (int)(rem % Nd);
    dst[l * plane + (long)n * Kd + k] = f2bf(src[i]);
  }
}

// ---- embedding ------------------------------------------------------------

__global__ void embed_kernel(const int* __restrict__ idx,
                             const float* __restrict__ tok,
                             const float* __restrict__ pos,
                             float* __restrict__ x) {
  size_t total = (size_t)kBT * kD;
  size_t stride = (size_t)gridDim.x * blockDim.x;
  for (size_t i = (size_t)blockIdx.x * blockDim.x + threadIdx.x; i < total;
       i += stride) {
    int bt = (int)(i / kD);
    int d  = (int)(i % kD);
    int tt = bt % kT;
    x[i] = tok[(size_t)idx[bt] * kD + d] + pos[(size_t)tt * kD + d];
  }
}

// ---- layernorm (fp32 in -> bf16 out) --------------------------------------

__global__ __launch_bounds__(256) void layernorm_kernel(
    const float* __restrict__ x, const float* __restrict__ w,
    const float* __restrict__ bb, u16* __restrict__ out) {
  const int row = blockIdx.x;
  const float* xr = x + (size_t)row * kD;
  float s = 0.f, ss = 0.f;
  for (int i = threadIdx.x; i < kD; i += 256) {
    float v = xr[i];
    s += v;
    ss += v * v;
  }
#pragma unroll
  for (int m = 16; m >= 1; m >>= 1) {
    s  += __shfl_xor(s, m, 32);
    ss += __shfl_xor(ss, m, 32);
  }
  __shared__ float sh[8], sh2[8];
  int lane = threadIdx.x & 31, wid = threadIdx.x >> 5;
  if (lane == 0) { sh[wid] = s; sh2[wid] = ss; }
  __syncthreads();
  if (threadIdx.x == 0) {
    float ts = 0.f, tss = 0.f;
#pragma unroll
    for (int k = 0; k < 8; ++k) { ts += sh[k]; tss += sh2[k]; }
    sh[0] = ts; sh2[0] = tss;
  }
  __syncthreads();
  float mean = sh[0] * (1.f / kD);
  float var  = sh2[0] * (1.f / kD) - mean * mean;
  float inv  = rsqrtf(var + 1e-5f);
  for (int i = threadIdx.x; i < kD; i += 256)
    out[(size_t)row * kD + i] = f2bf((xr[i] - mean) * inv * w[i] + bb[i]);
}

// ---- WMMA GEMM ------------------------------------------------------------
// C[M,N] = A[M,K] x Bt[N,K]^T, bf16 in, f32 accumulate.
// 256 threads = 8 waves (2x4); block tile 128x256; wave tile 64x64
// (16 accumulators -> 16 WMMA per 32-deep K step). LDS double buffered,
// async-copy staged: tile i+1 streams in while tile i feeds the WMMA pipe.
// mode 0: bf16 out | 1: bf16 out + bias + ReLU | 2: f32 out + bias + residual

__global__ __launch_bounds__(256) void gemm_bf16_kernel(
    const u16* __restrict__ A, const u16* __restrict__ Bt,
    const float* __restrict__ bias, const float* __restrict__ resid,
    void* __restrict__ outp, int M, int N, int K, int mode) {
  __shared__ __align__(16) u16 ldsA[2][128 * 32];   // [m][k]
  __shared__ __align__(16) u16 ldsB[2][256 * 32];   // [n][k]

  const int t = threadIdx.x;
  const int lane = t & 31, wid = t >> 5;
  const int waveM = wid & 1, waveN = wid >> 1;       // 2 x 4 wave grid
  const int blockN = blockIdx.x * 256, blockM = blockIdx.y * 128;
  const int l16 = lane & 15;

  v8f acc[4][4];
  {
    v8f z = v8f_zero();
#pragma unroll
    for (int mi = 0; mi < 4; ++mi)
#pragma unroll
      for (int ni = 0; ni < 4; ++ni) acc[mi][ni] = z;
  }

  const int ar = t >> 1, ac = (t & 1) * 16;   // A loader: 2 x 16B per thread
  const int br = t;                           // B loader: row t, 4 x 16B

  auto stage = [&](int kt, int buf) {
    const u16* ag = &A[(size_t)(blockM + ar) * K + kt + ac];
    u16* la = &ldsA[buf][ar * 32 + ac];
    cp16(la, ag);
    cp16(la + 8, ag + 8);
    const u16* bg = &Bt[(size_t)(blockN + br) * K + kt];
    u16* lb = &ldsB[buf][br * 32];
    cp16(lb, bg);
    cp16(lb + 8, bg + 8);
    cp16(lb + 16, bg + 16);
    cp16(lb + 24, bg + 24);
  };

  const int nk = K / 32;
  stage(0, 0);
  for (int ik = 0; ik < nk; ++ik) {
    const int buf = ik & 1;
    if (ik + 1 < nk) {
      stage((ik + 1) * 32, buf ^ 1);   // prefetch next tile into other buffer
      WAIT_ASYNC(6);                   // this wave's 6 ops for tile ik retired
    } else {
      WAIT_ASYNC(0);
    }
    __syncthreads();

    v16bf af[4], bf[4];
#pragma unroll
    for (int mi = 0; mi < 4; ++mi)
      af[mi] = frag_a(ldsA[buf], waveM * 64 + mi * 16 + l16, 32, lane, 0);
#pragma unroll
    for (int ni = 0; ni < 4; ++ni)
      bf[ni] = frag_b(ldsB[buf], waveN * 64 + ni * 16 + l16, 32, lane, 0);
#pragma unroll
    for (int mi = 0; mi < 4; ++mi)
#pragma unroll
      for (int ni = 0; ni < 4; ++ni)
        acc[mi][ni] = wmma_bf16(af[mi], bf[ni], acc[mi][ni]);

    __syncthreads();   // all waves done with buf before it is restaged
  }

  // epilogue: C layout — VGPR r: lanes 0-15 -> row r, lanes 16-31 -> row r+8
  const int rhalf = (lane & 16) ? 8 : 0;
#pragma unroll
  for (int mi = 0; mi < 4; ++mi) {
#pragma unroll
    for (int ni = 0; ni < 4; ++ni) {
#pragma unroll
      for (int r = 0; r < 8; ++r) {
        int row = blockM + waveM * 64 + mi * 16 + r + rhalf;
        int col = blockN + waveN * 64 + ni * 16 + l16;
        size_t off = (size_t)row * N + col;
        float v = acc[mi][ni][r];
        if (mode == 0) {
          ((u16*)outp)[off] = f2bf(v);
        } else if (mode == 1) {
          v = fmaxf(v + bias[col], 0.f);
          ((u16*)outp)[off] = f2bf(v);
        } else {
          ((float*)outp)[off] = v + bias[col] + resid[off];
        }
      }
    }
  }
}

// ---- flash attention ------------------------------------------------------
// Grid: (T/64, H, B). Block: 128 threads = 4 waves; each wave owns 16 query
// rows. KV streamed in 64-wide tiles; online softmax; S and PV via WMMA.

__global__ __launch_bounds__(128) void flash_attn_kernel(
    const u16* __restrict__ Q, const u16* __restrict__ Kc,
    const u16* __restrict__ Vc, u16* __restrict__ O, float scale) {
  __shared__ __align__(16) u16 ldsK[64 * 64];       // [seq][hd]
  __shared__ __align__(16) u16 ldsV[64 * 64];       // transposed [hd][seq]
  __shared__ __align__(16) u16 ldsP[4 * 16 * 64];   // per-wave P tiles

  const int t = threadIdx.x, lane = t & 31, wave = t >> 5;
  const int i = blockIdx.x, h = blockIdx.y, b = blockIdx.z;
  const int l16 = lane & 15;
  const int rhalf = (lane & 16) ? 8 : 0;

  // this wave's Q fragments (16 rows x 64) straight from global
  v16bf aQ[2];
  {
    const int qrow = i * 64 + wave * 16 + l16;
    const u16* qp = Q + ((size_t)(b * kT + qrow)) * kD + h * kHD;
#pragma unroll
    for (int c = 0; c < 2; ++c) {
      int k0 = c * 32 + ((lane & 16) ? 8 : 0);
      FragBF f;
      f.q[0] = *(const uint4*)(qp + k0);
      f.q[1] = *(const uint4*)(qp + k0 + 16);
      aQ[c] = f.v;
    }
  }

  v8f oacc[4];
  {
    v8f z = v8f_zero();
#pragma unroll
    for (int tt = 0; tt < 4; ++tt) oacc[tt] = z;
  }
  float rmax[8], rsum[8];
#pragma unroll
  for (int r = 0; r < 8; ++r) { rmax[r] = -1e30f; rsum[r] = 0.f; }

  const int kr = t >> 1, khalf = (t & 1) * 32;      // KV tile loader coords

  for (int j = 0; j <= i; ++j) {
    __syncthreads();
    {
      size_t goff = ((size_t)(b * kT + j * 64 + kr)) * kD + h * kHD + khalf;
      // K tile: async copy straight into LDS (consumed row-major)
      const u16* ks = Kc + goff;
      u16* kd = &ldsK[kr * 64 + khalf];
      cp16(kd, ks);
      cp16(kd + 8, ks + 8);
      cp16(kd + 16, ks + 16);
      cp16(kd + 24, ks + 24);
      // V tile: needs transpose -> VGPR staged scatter
      union { uint4 q[4]; u16 s[32]; } f;
      const uint4* vs = (const uint4*)(Vc + goff);
      f.q[0] = vs[0]; f.q[1] = vs[1]; f.q[2] = vs[2]; f.q[3] = vs[3];
#pragma unroll
      for (int e = 0; e < 32; ++e) ldsV[(khalf + e) * 64 + kr] = f.s[e];
    }
    WAIT_ASYNC(0);
    __syncthreads();

    // S = Q K^T  (B operand = K tile: n = seq pos, k = hd contiguous)
    v8f sacc[4];
    {
      v8f z = v8f_zero();
#pragma unroll
      for (int ni = 0; ni < 4; ++ni) sacc[ni] = z;
    }
#pragma unroll
    for (int c = 0; c < 2; ++c)
#pragma unroll
      for (int ni = 0; ni < 4; ++ni) {
        v16bf bk = frag_b(ldsK, ni * 16 + l16, 64, lane, c * 32);
        sacc[ni] = wmma_bf16(aQ[c], bk, sacc[ni]);
      }

    // online softmax
    float sv[4][8];
    const int rowg0 = i * 64 + wave * 16 + rhalf;
#pragma unroll
    for (int ni = 0; ni < 4; ++ni) {
      int col = j * 64 + ni * 16 + l16;
#pragma unroll
      for (int r = 0; r < 8; ++r) {
        float s = sacc[ni][r] * scale;
        if (col > rowg0 + r) s = -1e30f;   // causal mask
        sv[ni][r] = s;
      }
    }
#pragma unroll
    for (int r = 0; r < 8; ++r) {
      float tm = fmaxf(fmaxf(sv[0][r], sv[1][r]), fmaxf(sv[2][r], sv[3][r]));
      tm = redmax16(tm);
      float nm = fmaxf(rmax[r], tm);
      float corr = __expf(rmax[r] - nm);
      float ps = 0.f;
#pragma unroll
      for (int ni = 0; ni < 4; ++ni) {
        float pp = __expf(sv[ni][r] - nm);
        sv[ni][r] = pp;
        ps += pp;
      }
      ps = redsum16(ps);
      rsum[r] = rsum[r] * corr + ps;
      rmax[r] = nm;
#pragma unroll
      for (int tt = 0; tt < 4; ++tt) oacc[tt][r] *= corr;
    }

    // park P in this wave's private LDS region, reload as A fragments
    u16* pw = &ldsP[wave * 1024];
#pragma unroll
    for (int ni = 0; ni < 4; ++ni)
#pragma unroll
      for (int r = 0; r < 8; ++r)
        pw[(r + rhalf) * 64 + ni * 16 + l16] = f2bf(sv[ni][r]);

    // O += P V   (B operand = V^T tile: n = hd, k = seq contiguous)
#pragma unroll
    for (int c = 0; c < 2; ++c) {
      v16bf aP = frag_a(pw, l16, 64, lane, c * 32);
#pragma unroll
      for (int tt = 0; tt < 4; ++tt) {
        v16bf bv = frag_b(ldsV, tt * 16 + l16, 64, lane, c * 32);
        oacc[tt] = wmma_bf16(aP, bv, oacc[tt]);
      }
    }
  }

  // normalize and write context (layout [B,T,H*HD] == [BT, D])
#pragma unroll
  for (int tt = 0; tt < 4; ++tt) {
#pragma unroll
    for (int r = 0; r < 8; ++r) {
      int row = i * 64 + wave * 16 + rhalf + r;
      int col = h * kHD + tt * 16 + l16;
      O[((size_t)(b * kT + row)) * kD + col] = f2bf(oacc[tt][r] / rsum[r]);
    }
  }
}

// ---------------------------------------------------------------------------

extern "C" void kernel_launch(void* const* d_in, const int* in_sizes, int n_in,
                              void* d_out, int out_size, void* d_ws,
                              size_t ws_size, hipStream_t stream) {
  const int*   idx    = (const int*)d_in[0];
  const float* tok    = (const float*)d_in[1];
  const float* pos    = (const float*)d_in[2];
  const float* wq     = (const float*)d_in[3];
  const float* wk     = (const float*)d_in[4];
  const float* wv     = (const float*)d_in[5];
  const float* wproj  = (const float*)d_in[6];
  const float* bproj  = (const float*)d_in[7];
  const float* ln1w   = (const float*)d_in[8];
  const float* ln1b   = (const float*)d_in[9];
  const float* ln2w   = (const float*)d_in[10];
  const float* ln2b   = (const float*)d_in[11];
  const float* w1     = (const float*)d_in[12];
  const float* b1     = (const float*)d_in[13];
  const float* w2     = (const float*)d_in[14];
  const float* b2     = (const float*)d_in[15];

  // workspace carve (~272 MB total)
  char* p = (char*)d_ws;
  auto carve = [&](size_t bytes) -> char* {
    char* r = p;
    p += (bytes + 255) & ~(size_t)255;
    return r;
  };
  const size_t nW_dd  = (size_t)kL * kD * kD;
  const size_t nW_dff = (size_t)kL * kD * kDF;

  u16*   wq_bf  = (u16*)carve(nW_dd * 2);    // all weight stacks stored [N,K]
  u16*   wk_bf  = (u16*)carve(nW_dd * 2);
  u16*   wv_bf  = (u16*)carve(nW_dd * 2);
  u16*   wp_bf  = (u16*)carve(nW_dd * 2);
  u16*   w1_bf  = (u16*)carve(nW_dff * 2);
  u16*   w2_bf  = (u16*)carve(nW_dff * 2);
  float* x      = (float*)carve((size_t)kBT * kD * 4);
  u16*   h_bf   = (u16*)carve((size_t)kBT * kD * 2);
  u16*   q_bf   = (u16*)carve((size_t)kBT * kD * 2);
  u16*   k_bf   = (u16*)carve((size_t)kBT * kD * 2);
  u16*   v_bf   = (u16*)carve((size_t)kBT * kD * 2);
  u16*   ctx_bf = (u16*)carve((size_t)kBT * kD * 2);
  u16*   ff_bf  = (u16*)carve((size_t)kBT * kDF * 2);

  // 1) weights -> bf16, transposed to [N,K] (deterministic, every call)
  convert_transpose_kernel<<<4096, 256, 0, stream>>>(wq, wq_bf, kD, kD,
                                                     (long)nW_dd);
  convert_transpose_kernel<<<4096, 256, 0, stream>>>(wk, wk_bf, kD, kD,
                                                     (long)nW_dd);
  convert_transpose_kernel<<<4096, 256, 0, stream>>>(wv, wv_bf, kD, kD,
                                                     (long)nW_dd);
  convert_transpose_kernel<<<4096, 256, 0, stream>>>(wproj, wp_bf, kD, kD,
                                                     (long)nW_dd);
  convert_transpose_kernel<<<4096, 256, 0, stream>>>(w1, w1_bf, kD, kDF,
                                                     (long)nW_dff);
  convert_transpose_kernel<<<4096, 256, 0, stream>>>(w2, w2_bf, kDF, kD,
                                                     (long)nW_dff);

  // 2) embeddings
  embed_kernel<<<4096, 256, 0, stream>>>(idx, tok, pos, x);

  const float scale = rsqrtf((float)kD);   // reference scales by D^-0.5
  const dim3 gemmBlk(256);
  const dim3 gD(kD / 256, kBT / 128);      // N=1024 outputs
  const dim3 gDF(kDF / 256, kBT / 128);    // N=4096 outputs
  const dim3 gAttn(kT / 64, kH, kB);

  for (int l = 0; l < kL; ++l) {
    const size_t odd  = (size_t)l * kD * kD;
    const size_t odff = (size_t)l * kD * kDF;

    layernorm_kernel<<<kBT, 256, 0, stream>>>(x, ln1w + l * kD, ln1b + l * kD,
                                              h_bf);
    gemm_bf16_kernel<<<gD, gemmBlk, 0, stream>>>(h_bf, wq_bf + odd, nullptr,
                                                 nullptr, q_bf, kBT, kD, kD, 0);
    gemm_bf16_kernel<<<gD, gemmBlk, 0, stream>>>(h_bf, wk_bf + odd, nullptr,
                                                 nullptr, k_bf, kBT, kD, kD, 0);
    gemm_bf16_kernel<<<gD, gemmBlk, 0, stream>>>(h_bf, wv_bf + odd, nullptr,
                                                 nullptr, v_bf, kBT, kD, kD, 0);
    flash_attn_kernel<<<gAttn, 128, 0, stream>>>(q_bf, k_bf, v_bf, ctx_bf,
                                                 scale);
    gemm_bf16_kernel<<<gD, gemmBlk, 0, stream>>>(ctx_bf, wp_bf + odd,
                                                 bproj + l * kD, x, x, kBT, kD,
                                                 kD, 2);
    layernorm_kernel<<<kBT, 256, 0, stream>>>(x, ln2w + l * kD, ln2b + l * kD,
                                              h_bf);
    gemm_bf16_kernel<<<gDF, gemmBlk, 0, stream>>>(h_bf, w1_bf + odff,
                                                  b1 + l * kDF, nullptr, ff_bf,
                                                  kBT, kDF, kD, 1);
    gemm_bf16_kernel<<<gD, gemmBlk, 0, stream>>>(ff_bf, w2_bf + odff,
                                                 b2 + l * kD, x, x, kBT, kD,
                                                 kDF, 2);
  }

  (void)hipMemcpyAsync(d_out, x, (size_t)kBT * kD * sizeof(float),
                       hipMemcpyDeviceToDevice, stream);
}
